// IrisTransformer_20796231647762
// MI455X (gfx1250) — matrix operand order repaired and checked by reading, not compiled, not run on hardware
//
#include <hip/hip_runtime.h>
#include <math.h>

// ---------------- problem constants (match reference) ----------------
constexpr int L_   = 10;
constexpr int B_   = 2;
constexpr int T_   = 2040;
constexpr int D_   = 256;
constexpr int H_   = 4;
constexpr int HD_  = 64;     // head dim
constexpr int TPB_ = 17;     // tokens per block (mask)
constexpr int MTOK = B_ * T_;        // 4080 rows
constexpr int D4_  = 4 * D_;         // 1024

typedef __attribute__((ext_vector_type(16))) _Float16 v16h;
typedef __attribute__((ext_vector_type(8)))  _Float16 v8h;
typedef __attribute__((ext_vector_type(8)))  float    v8f;
typedef __attribute__((ext_vector_type(4)))  int      v4i;

// address-space qualified v4i pointers for the async builtin
typedef __attribute__((address_space(1))) v4i as1_v4i;
typedef __attribute__((address_space(3))) v4i as3_v4i;

__device__ inline v16h join16(v8h lo, v8h hi) {
  v16h r;
#pragma unroll
  for (int i = 0; i < 8; ++i) { r[i] = lo[i]; r[i + 8] = hi[i]; }
  return r;
}

// ---------------- CDNA5 async global->LDS (guarded; falls back to VGPR copy) ----
#if defined(__HIP_DEVICE_COMPILE__) && __has_builtin(__builtin_amdgcn_global_load_async_to_lds_b128)
#define HAVE_ASYNC_LDS 1
#else
#define HAVE_ASYNC_LDS 0
#endif

__device__ inline void async_copy_b128(const _Float16* gsrc, _Float16* ldst) {
#if HAVE_ASYNC_LDS
  __builtin_amdgcn_global_load_async_to_lds_b128(
      (as1_v4i*)(void*)(_Float16*)gsrc,
      (as3_v4i*)(void*)ldst, 0, 0);
#else
  *(v8h*)ldst = *(const v8h*)gsrc;
#endif
}

__device__ inline void wait_async_lds() {
#if HAVE_ASYNC_LDS && __has_builtin(__builtin_amdgcn_s_wait_asynccnt)
  __builtin_amdgcn_s_wait_asynccnt(0);
#endif
}

// ---------------- fp32 -> fp16 cast (weights) ----------------
__global__ void f32_to_f16_kernel(const float* __restrict__ src,
                                  _Float16* __restrict__ dst, long n) {
  long i = (long)blockIdx.x * blockDim.x + threadIdx.x;
  long stride = (long)gridDim.x * blockDim.x;
  for (; i < n; i += stride) dst[i] = (_Float16)src[i];
}

// ---------------- LayerNorm: one 256-thread block per row ----------------
// mode 0: write f16 to out16; mode 1: write f32 to out32
__global__ __launch_bounds__(256) void ln_kernel(
    const float* __restrict__ x, const float* __restrict__ g,
    const float* __restrict__ be, _Float16* __restrict__ out16,
    float* __restrict__ out32, int mode) {
  int row = blockIdx.x;
  int t = threadIdx.x;           // 0..255 == D
  __shared__ float red[256];
  float v = x[(long)row * D_ + t];
  red[t] = v;
  __syncthreads();
  for (int s = 128; s > 0; s >>= 1) {
    if (t < s) red[t] += red[t + s];
    __syncthreads();
  }
  float mu = red[0] * (1.0f / D_);
  __syncthreads();
  float dv = v - mu;
  red[t] = dv * dv;
  __syncthreads();
  for (int s = 128; s > 0; s >>= 1) {
    if (t < s) red[t] += red[t + s];
    __syncthreads();
  }
  float var = red[0] * (1.0f / D_);
  float r = dv * rsqrtf(var + 1e-5f) * g[t] + be[t];
  if (mode == 0) out16[(long)row * D_ + t] = (_Float16)r;
  else           out32[(long)row * D_ + t] = r;
}

// ---------------- WMMA GEMM: C = A(f16)[M,K] @ B(f16)[K,N] + bias ----------------
// 64x64 block tile, 8 waves (4 in M x 2 in N), each wave: 16x32 via 2 accums.
// mode 0: out16 = f16(acc+bias)
// mode 1: resio += acc+bias           (f32 residual, in place)
// mode 2: out16 = f16(gelu(acc+bias)) (exact erf gelu)
constexpr int BM = 64, BN = 64, BK = 32;
constexpr int LDA_S = 40;   // halfs; row stride 80B (16B aligned)
constexpr int LDBT_S = 40;  // transposed B: [col][k], stride 80B

__global__ __launch_bounds__(256) void gemm_wmma_kernel(
    const _Float16* __restrict__ A, const _Float16* __restrict__ Bw,
    const float* __restrict__ bias, _Float16* __restrict__ out16,
    float* __restrict__ resio, int M, int N, int K, int mode) {
  __shared__ __align__(16) _Float16 sA[BM * LDA_S];
  __shared__ __align__(16) _Float16 sBT[BN * LDBT_S];

  int tid = threadIdx.x;
  int lane = tid & 31;
  int wave = tid >> 5;      // 0..7
  int wm = wave & 3;        // 4 waves along M
  int wn = wave >> 2;       // 2 waves along N
  int l16 = lane & 15;
  int hi = lane >> 4;       // 0: lanes 0-15, 1: lanes 16-31
  int gm0 = blockIdx.x * BM;
  int gn0 = blockIdx.y * BN;

  // staging coordinates (fixed per thread)
  int ar = tid >> 2;                 // A tile row 0..63
  int ac = (tid & 3) * 8;            // A tile col chunk
  int agr = gm0 + ar; if (agr > M - 1) agr = M - 1;   // clamp: OOB rows never stored
  int br = tid >> 3;                 // B tile row 0..31
  int bc = (tid & 7) * 8;            // B tile col chunk

  v8f acc0, acc1;
#pragma unroll
  for (int r = 0; r < 8; ++r) { acc0[r] = 0.f; acc1[r] = 0.f; }

  for (int k0 = 0; k0 < K; k0 += BK) {
    // stage A tile [64 x 32] row-major via async global->LDS (16B per thread)
    async_copy_b128(A + (long)agr * K + k0 + ac, sA + ar * LDA_S + ac);

    // stage B tile [32 x 64] transposed into sBT[col][k]
    {
      v8h val = *(const v8h*)(Bw + (long)(k0 + br) * N + gn0 + bc);
#pragma unroll
      for (int j = 0; j < 8; ++j) sBT[(bc + j) * LDBT_S + br] = val[j];
    }
    // prefetch next K-step tiles into cache while this step computes
    if (k0 + BK < K) {
      __builtin_prefetch(A + (long)agr * K + k0 + BK + ac, 0, 0);
      __builtin_prefetch(Bw + (long)(k0 + BK + br) * N + gn0 + bc, 0, 0);
    }
    wait_async_lds();
    __syncthreads();

    // A fragment: lane = M (0-15), halfs: K kb..kb+7, 16+kb..16+kb+7
    int am = wm * 16 + l16;
    int kb = hi * 8;
    v8h a_lo = *(const v8h*)(sA + am * LDA_S + kb);
    v8h a_hi = *(const v8h*)(sA + am * LDA_S + 16 + kb);
    v16h af = join16(a_lo, a_hi);

    int kbb = hi * 8;    // B fragment: halfs K kbb..kbb+7, 16+kbb..16+kbb+7
#pragma unroll
    for (int t = 0; t < 2; ++t) {
      int col = wn * 32 + t * 16 + l16;
      v8h b_lo = *(const v8h*)(sBT + col * LDBT_S + kbb);
      v8h b_hi = *(const v8h*)(sBT + col * LDBT_S + kbb + 16);
      v16h bf = join16(b_lo, b_hi);
      if (t == 0)
        acc0 = __builtin_amdgcn_wmma_f32_16x16x32_f16(false, af, false, bf,
                                                      (short)0, acc0, false, false);
      else
        acc1 = __builtin_amdgcn_wmma_f32_16x16x32_f16(false, af, false, bf,
                                                      (short)0, acc1, false, false);
    }
    __syncthreads();
  }

  // epilogue: C layout lane = N, M = r + 8*hi
#pragma unroll
  for (int t = 0; t < 2; ++t) {
    int col = gn0 + wn * 32 + t * 16 + l16;
    float bz = bias ? bias[col] : 0.f;
#pragma unroll
    for (int r = 0; r < 8; ++r) {
      int row = gm0 + wm * 16 + r + hi * 8;
      if (row < M) {
        float v = (t == 0 ? acc0[r] : acc1[r]) + bz;
        long idx = (long)row * N + col;
        if (mode == 0) {
          out16[idx] = (_Float16)v;
        } else if (mode == 1) {
          resio[idx] = resio[idx] + v;
        } else {
          float gv = 0.5f * v * (1.0f + erff(v * 0.70710678118654752f));
          out16[idx] = (_Float16)gv;
        }
      }
    }
  }
}

// ---------------- Flash attention w/ block-causal mask ----------------
// One wave per (b, h, 16-query tile). Streams keys in tiles of 32.
constexpr int NQT = (T_ + 15) / 16;   // 128

__global__ __launch_bounds__(32) void attn_kernel(
    const _Float16* __restrict__ q, const _Float16* __restrict__ k,
    const _Float16* __restrict__ v, _Float16* __restrict__ y) {
  __shared__ __align__(16) _Float16 sK[32 * 72];   // [key][d], stride 144B
  __shared__ __align__(16) _Float16 sVT[64 * 40];  // [d][key], stride 80B
  __shared__ __align__(16) _Float16 sP[16 * 40];   // [m][key], stride 80B

  int id = blockIdx.x;
  int qt = id % NQT;
  int h  = (id / NQT) % H_;
  int b  = id / (NQT * H_);
  int lane = threadIdx.x;
  int l16 = lane & 15;
  int hi = lane >> 4;
  int q0 = qt * 16;

  // Q fragments (A layout): lane = query row, K-dim = head dim
  int qrow = q0 + l16;
  if (qrow >= T_) qrow = T_ - 1;
  const _Float16* qp = q + ((long)(b * T_ + qrow)) * D_ + h * HD_;
  int kb = hi * 8;
  v8h q0a = *(const v8h*)(qp + kb);
  v8h q0b = *(const v8h*)(qp + 16 + kb);
  v8h q1a = *(const v8h*)(qp + 32 + kb);
  v8h q1b = *(const v8h*)(qp + 48 + kb);
  v16h qf0 = join16(q0a, q0b);
  v16h qf1 = join16(q1a, q1b);

  float mrow[8], lrow[8];
  v8f O[4];
#pragma unroll
  for (int t = 0; t < 4; ++t)
#pragma unroll
    for (int r = 0; r < 8; ++r) O[t][r] = 0.f;
#pragma unroll
  for (int r = 0; r < 8; ++r) { mrow[r] = -1e30f; lrow[r] = 0.f; }

  int qlast = q0 + 15; if (qlast > T_ - 1) qlast = T_ - 1;
  int kend = (qlast / TPB_) * TPB_ + (TPB_ - 1);
  if (kend < qlast) kend = qlast;
  if (kend > T_ - 1) kend = T_ - 1;
  const float scale = 0.125f;   // 1/sqrt(64)

  for (int j0 = 0; j0 <= kend; j0 += 32) {
    // ---- stage K (row-major, async) and V (transposed) tiles; clamp OOB rows.
    // Clamped rows are finite and fully masked later (no 0*Inf hazards).
    int krow = j0 + lane; if (krow > T_ - 1) krow = T_ - 1;
    const _Float16* kp = k + ((long)(b * T_ + krow)) * D_ + h * HD_;
    const _Float16* vp = v + ((long)(b * T_ + krow)) * D_ + h * HD_;
#pragma unroll
    for (int c = 0; c < 8; ++c) {
      async_copy_b128(kp + c * 8, sK + lane * 72 + c * 8);
      v8h vvv = *(const v8h*)(vp + c * 8);
#pragma unroll
      for (int j = 0; j < 8; ++j) sVT[(c * 8 + j) * 40 + lane] = vvv[j];
    }
    wait_async_lds();
    __syncthreads();

    // ---- S = Q @ K^T for two 16-key subtiles
    v8f S[2];
#pragma unroll
    for (int jt = 0; jt < 2; ++jt) {
      int key = jt * 16 + l16;     // B-frag lane = key column
      int dbase = hi * 8;          // halfs K(=d) dbase..+7, 16+dbase..+7
      v16h kf0 = join16(*(const v8h*)(sK + key * 72 + dbase),
                        *(const v8h*)(sK + key * 72 + dbase + 16));
      v16h kf1 = join16(*(const v8h*)(sK + key * 72 + 32 + dbase),
                        *(const v8h*)(sK + key * 72 + 32 + dbase + 16));
      v8f s;
#pragma unroll
      for (int r = 0; r < 8; ++r) s[r] = 0.f;
      s = __builtin_amdgcn_wmma_f32_16x16x32_f16(false, qf0, false, kf0,
                                                 (short)0, s, false, false);
      s = __builtin_amdgcn_wmma_f32_16x16x32_f16(false, qf1, false, kf1,
                                                 (short)0, s, false, false);
      S[jt] = s;
    }

    // ---- mask + online softmax (rows striped: M = r + 8*hi, lane = key%16)
#pragma unroll
    for (int r = 0; r < 8; ++r) {
      int qi = q0 + r + hi * 8;
      int g0 = j0 + l16;
      int g1 = j0 + 16 + l16;
      bool ok0 = (g0 < T_) && (qi < T_) &&
                 ((g0 <= qi) || (g0 / TPB_ == qi / TPB_));
      bool ok1 = (g1 < T_) && (qi < T_) &&
                 ((g1 <= qi) || (g1 / TPB_ == qi / TPB_));
      float s0v = ok0 ? S[0][r] * scale : -1e30f;
      float s1v = ok1 ? S[1][r] * scale : -1e30f;
      float mx = fmaxf(s0v, s1v);
      mx = fmaxf(mx, __shfl_xor(mx, 1));
      mx = fmaxf(mx, __shfl_xor(mx, 2));
      mx = fmaxf(mx, __shfl_xor(mx, 4));
      mx = fmaxf(mx, __shfl_xor(mx, 8));
      float mnew = fmaxf(mrow[r], mx);
      float a = __expf(mrow[r] - mnew);
      float p0 = __expf(s0v - mnew);
      float p1 = __expf(s1v - mnew);
      float sum = p0 + p1;
      sum += __shfl_xor(sum, 1);
      sum += __shfl_xor(sum, 2);
      sum += __shfl_xor(sum, 4);
      sum += __shfl_xor(sum, 8);
      lrow[r] = lrow[r] * a + sum;
      mrow[r] = mnew;
      sP[(r + hi * 8) * 40 + l16]      = (_Float16)p0;
      sP[(r + hi * 8) * 40 + 16 + l16] = (_Float16)p1;
#pragma unroll
      for (int t = 0; t < 4; ++t) O[t][r] *= a;
    }
    __syncthreads();

    // ---- P fragment (A layout: lane = query row, K = 32 keys)
    v16h pf = join16(*(const v8h*)(sP + l16 * 40 + kb),
                     *(const v8h*)(sP + l16 * 40 + 16 + kb));

    // ---- O += P @ V, four 16-wide d subtiles
#pragma unroll
    for (int t = 0; t < 4; ++t) {
      int dcol = t * 16 + l16;     // B-frag lane = output d column
      v16h vf = join16(*(const v8h*)(sVT + dcol * 40 + hi * 8),
                       *(const v8h*)(sVT + dcol * 40 + hi * 8 + 16));
      O[t] = __builtin_amdgcn_wmma_f32_16x16x32_f16(false, pf, false, vf,
                                                    (short)0, O[t], false, false);
    }
    __syncthreads();
  }

  // ---- normalize + store y (f16, [B,T,D] with head columns)
#pragma unroll
  for (int r = 0; r < 8; ++r) {
    int row = q0 + r + hi * 8;
    if (row < T_) {
      float inv = lrow[r] > 0.f ? 1.0f / lrow[r] : 0.f;
      long base = ((long)(b * T_ + row)) * D_ + h * HD_;
      y[base +      l16] = (_Float16)(O[0][r] * inv);
      y[base + 16 + l16] = (_Float16)(O[1][r] * inv);
      y[base + 32 + l16] = (_Float16)(O[2][r] * inv);
      y[base + 48 + l16] = (_Float16)(O[3][r] * inv);
    }
  }
}

// ---------------- host orchestration ----------------
extern "C" void kernel_launch(void* const* d_in, const int* in_sizes, int n_in,
                              void* d_out, int out_size, void* d_ws, size_t ws_size,
                              hipStream_t stream) {
  const float* seq    = (const float*)d_in[0];
  const float* ln1_g  = (const float*)d_in[1];
  const float* ln1_b  = (const float*)d_in[2];
  const float* wq     = (const float*)d_in[3];
  const float* bq     = (const float*)d_in[4];
  const float* wk     = (const float*)d_in[5];
  const float* bk     = (const float*)d_in[6];
  const float* wv     = (const float*)d_in[7];
  const float* bv     = (const float*)d_in[8];
  const float* wproj  = (const float*)d_in[9];
  const float* bproj  = (const float*)d_in[10];
  const float* ln2_g  = (const float*)d_in[11];
  const float* ln2_b  = (const float*)d_in[12];
  const float* w1     = (const float*)d_in[13];
  const float* b1     = (const float*)d_in[14];
  const float* w2     = (const float*)d_in[15];
  const float* b2     = (const float*)d_in[16];
  const float* lnf_g  = (const float*)d_in[17];
  const float* lnf_b  = (const float*)d_in[18];

  char* ws = (char*)d_ws;
  size_t off = 0;
  auto take = [&](size_t bytes) -> void* {
    void* p = (void*)(ws + off);
    off += (bytes + 255) & ~(size_t)255;
    return p;
  };

  float*     x    = (float*)take((size_t)MTOK * D_ * 4);
  _Float16*  h16  = (_Float16*)take((size_t)MTOK * D_ * 2);
  _Float16*  q16  = (_Float16*)take((size_t)MTOK * D_ * 2);
  _Float16*  k16  = (_Float16*)take((size_t)MTOK * D_ * 2);
  _Float16*  v16  = (_Float16*)take((size_t)MTOK * D_ * 2);
  _Float16*  y16  = (_Float16*)take((size_t)MTOK * D_ * 2);
  _Float16*  g16  = (_Float16*)take((size_t)MTOK * D4_ * 2);
  _Float16*  wq16 = (_Float16*)take((size_t)L_ * D_ * D_ * 2);
  _Float16*  wk16 = (_Float16*)take((size_t)L_ * D_ * D_ * 2);
  _Float16*  wv16 = (_Float16*)take((size_t)L_ * D_ * D_ * 2);
  _Float16*  wp16 = (_Float16*)take((size_t)L_ * D_ * D_ * 2);
  _Float16*  w116 = (_Float16*)take((size_t)L_ * D_ * D4_ * 2);
  _Float16*  w216 = (_Float16*)take((size_t)L_ * D4_ * D_ * 2);

  // x <- sequences (fp32 residual stream)
  (void)hipMemcpyAsync(x, seq, (size_t)MTOK * D_ * 4, hipMemcpyDeviceToDevice, stream);

  // cast all weights once
  auto cast = [&](const float* s, _Float16* d, long n) {
    int blocks = (int)((n + 255) / 256);
    if (blocks > 4096) blocks = 4096;
    f32_to_f16_kernel<<<blocks, 256, 0, stream>>>(s, d, n);
  };
  cast(wq, wq16, (long)L_ * D_ * D_);
  cast(wk, wk16, (long)L_ * D_ * D_);
  cast(wv, wv16, (long)L_ * D_ * D_);
  cast(wproj, wp16, (long)L_ * D_ * D_);
  cast(w1, w116, (long)L_ * D_ * D4_);
  cast(w2, w216, (long)L_ * D4_ * D_);

  const int gm = (MTOK + BM - 1) / BM;   // 64
  dim3 gD(gm, D_ / BN);                  // (64, 4)
  dim3 gD4(gm, D4_ / BN);                // (64, 16)

  for (int l = 0; l < L_; ++l) {
    long wOff = (long)l * D_ * D_;
    long w1Off = (long)l * D_ * D4_;
    long w2Off = (long)l * D4_ * D_;

    // h = LN1(x) -> f16
    ln_kernel<<<MTOK, 256, 0, stream>>>(x, ln1_g + l * D_, ln1_b + l * D_,
                                        h16, nullptr, 0);
    // q,k,v = h @ W + b -> f16
    gemm_wmma_kernel<<<gD, 256, 0, stream>>>(h16, wq16 + wOff, bq + l * D_,
                                             q16, nullptr, MTOK, D_, D_, 0);
    gemm_wmma_kernel<<<gD, 256, 0, stream>>>(h16, wk16 + wOff, bk + l * D_,
                                             k16, nullptr, MTOK, D_, D_, 0);
    gemm_wmma_kernel<<<gD, 256, 0, stream>>>(h16, wv16 + wOff, bv + l * D_,
                                             v16, nullptr, MTOK, D_, D_, 0);
    // y = block-causal attention(q,k,v) -> f16
    attn_kernel<<<B_ * H_ * NQT, 32, 0, stream>>>(q16, k16, v16, y16);
    // x += y @ Wp + bp
    gemm_wmma_kernel<<<gD, 256, 0, stream>>>(y16, wp16 + wOff, bproj + l * D_,
                                             nullptr, x, MTOK, D_, D_, 1);
    // h = LN2(x) -> f16
    ln_kernel<<<MTOK, 256, 0, stream>>>(x, ln2_g + l * D_, ln2_b + l * D_,
                                        h16, nullptr, 0);
    // g = gelu(h @ W1 + b1) -> f16
    gemm_wmma_kernel<<<gD4, 256, 0, stream>>>(h16, w116 + w1Off, b1 + l * D4_,
                                              g16, nullptr, MTOK, D4_, D_, 2);
    // x += g @ W2 + b2
    gemm_wmma_kernel<<<gD, 256, 0, stream>>>(g16, w216 + w2Off, b2 + l * D_,
                                             nullptr, x, MTOK, D_, D4_, 1);
  }

  // out = LN_f(x) -> f32
  ln_kernel<<<MTOK, 256, 0, stream>>>(x, lnf_g, lnf_b, nullptr, (float*)d_out, 1);
}
